// DecoderBlock_51573967290862
// MI455X (gfx1250) — compile-verified, matrix operand order checked by reading, not run on hardware
//
#include <hip/hip_runtime.h>
#include <math.h>

// ---------------------------------------------------------------------------
// Types for CDNA5 WMMA (wave32)
// ---------------------------------------------------------------------------
typedef __attribute__((ext_vector_type(16))) _Float16 v16h;
typedef __attribute__((ext_vector_type(8)))  _Float16 v8h;
typedef __attribute__((ext_vector_type(4)))  _Float16 v4h;
typedef __attribute__((ext_vector_type(8)))  float    v8f;
typedef __attribute__((ext_vector_type(4)))  float    v4f;

union F16x16 { v16h v; v8h h[2]; };

__device__ __forceinline__ v8f zero_v8f() {
    v8f z;
#pragma unroll
    for (int i = 0; i < 8; ++i) z[i] = 0.0f;
    return z;
}

// ---------------------------------------------------------------------------
// LayerNorm: one row (384 elems) per 128-thread block
// ---------------------------------------------------------------------------
__global__ __launch_bounds__(128) void layernorm_k(
    const float* __restrict__ X, const float* __restrict__ gam,
    const float* __restrict__ bet, float* __restrict__ Y)
{
    __shared__ float red[2][4];
    const int tid = threadIdx.x;
    const size_t row = blockIdx.x;
    const float* xr = X + row * 384;

    float v0 = xr[tid], v1 = xr[tid + 128], v2 = xr[tid + 256];
    float s  = v0 + v1 + v2;
    float sq = v0 * v0 + v1 * v1 + v2 * v2;
#pragma unroll
    for (int off = 16; off > 0; off >>= 1) {
        s  += __shfl_xor(s,  off, 32);
        sq += __shfl_xor(sq, off, 32);
    }
    const int wave = tid >> 5, lane = tid & 31;
    if (lane == 0) { red[0][wave] = s; red[1][wave] = sq; }
    __syncthreads();
    s  = red[0][0] + red[0][1] + red[0][2] + red[0][3];
    sq = red[1][0] + red[1][1] + red[1][2] + red[1][3];

    const float mu  = s * (1.0f / 384.0f);
    const float var = sq * (1.0f / 384.0f) - mu * mu;
    const float inv = rsqrtf(var + 1e-5f);

    float* yr = Y + row * 384;
    yr[tid]       = (v0 - mu) * inv * gam[tid]       + bet[tid];
    yr[tid + 128] = (v1 - mu) * inv * gam[tid + 128] + bet[tid + 128];
    yr[tid + 256] = (v2 - mu) * inv * gam[tid + 256] + bet[tid + 256];
}

// ---------------------------------------------------------------------------
// Tiled f16-WMMA GEMM, fully compile-time shaped.
//   C[M,N] = A[M,K] @ B (+bias) (+residual) (opt ReLU); block tile 64x64.
//   B element (k, gn) at B[k*BSK + (gn>>6)*BSH + (gn&63)]:
//     plain [K,N] :        BSK = N,  BSH = 64
//     headed [H,C,D=64]:   BSK = 64, BSH = C*64
// ---------------------------------------------------------------------------
template<int N, int K, int BSK, int BSH, int HAS_BIAS, int HAS_RES, int RELU>
__global__ __launch_bounds__(128) void gemm_wmma_k(
    const float* __restrict__ A, const float* __restrict__ B,
    const float* __restrict__ bias, const float* __restrict__ resid,
    float* __restrict__ C)
{
    __shared__ _Float16 As[64][40];   // [m][k], 80B row stride
    __shared__ _Float16 Bs[64][40];   // [n][k]

    const int tid  = threadIdx.x;
    const int wave = tid >> 5;
    const int lane = tid & 31;
    const int ln15 = lane & 15;
    const int hi   = (lane >> 4) & 1;
    const int row0 = blockIdx.x * 64;
    const int col0 = blockIdx.y * 64;

    // A staging map: 4 float4 per thread, pointers hoisted out of the k loop
    const float* aP[4];
    v4h* aS[4];
#pragma unroll
    for (int i = 0; i < 4; ++i) {
        int e  = i * 128 + tid;
        int r  = e >> 3;               // 8 float4 per 32-wide row
        int c4 = e & 7;
        aP[i] = A + (size_t)(row0 + r) * K + c4 * 4;
        aS[i] = (v4h*)&As[r][c4 * 4];
    }
    // B staging map: one 4x4 (k x n) patch per thread, transposed into LDS
    const int bn  = (tid & 15) * 4;
    const int bk  = (tid >> 4) * 4;
    const int gbn = col0 + bn;
    const float* bP = B + (size_t)bk * BSK +
                      (size_t)(gbn >> 6) * BSH + (gbn & 63);

    v8f acc[4];
#pragma unroll
    for (int t = 0; t < 4; ++t) acc[t] = zero_v8f();

    for (int k0 = 0; k0 < K; k0 += 32) {
        // batched global loads into registers (single wait point)
        v4f av[4], bv[4];
#pragma unroll
        for (int i = 0; i < 4; ++i) av[i] = *(const v4f*)aP[i];
#pragma unroll
        for (int i = 0; i < 4; ++i) bv[i] = *(const v4f*)(bP + i * BSK);
#pragma unroll
        for (int i = 0; i < 4; ++i) aP[i] += 32;
        bP += 32 * BSK;
        __builtin_prefetch(aP[0], 0, 1);
        __builtin_prefetch(bP, 0, 1);

        __syncthreads();               // previous iteration's LDS reads done
#pragma unroll
        for (int i = 0; i < 4; ++i) {
            v4h hv;
#pragma unroll
            for (int j = 0; j < 4; ++j) hv[j] = (_Float16)av[i][j];
            *aS[i] = hv;
        }
#pragma unroll
        for (int j = 0; j < 4; ++j) {  // 4x4 register transpose for Bs[n][k]
            v4h hv;
#pragma unroll
            for (int i = 0; i < 4; ++i) hv[i] = (_Float16)bv[i][j];
            *(v4h*)&Bs[bn + j][bk] = hv;
        }
        __syncthreads();

        const int kb = hi * 8;
        F16x16 afrag, bfrag[4];
        afrag.h[0] = *(const v8h*)&As[wave * 16 + ln15][kb];
        afrag.h[1] = *(const v8h*)&As[wave * 16 + ln15][kb + 16];
#pragma unroll
        for (int t = 0; t < 4; ++t) {
            bfrag[t].h[0] = *(const v8h*)&Bs[t * 16 + ln15][kb];
            bfrag[t].h[1] = *(const v8h*)&Bs[t * 16 + ln15][kb + 16];
        }
#pragma unroll
        for (int t = 0; t < 4; ++t)
            acc[t] = __builtin_amdgcn_wmma_f32_16x16x32_f16(
                false, afrag.v, false, bfrag[t].v, (short)0, acc[t], false, false);
    }

    // epilogue: single base pointer per tile, constant row offsets (r*N)
    const int gm0 = row0 + wave * 16 + hi * 8;
    const size_t rowbase = (size_t)gm0 * N;
#pragma unroll
    for (int t = 0; t < 4; ++t) {
        const int gn = col0 + t * 16 + ln15;
        const float bvs = HAS_BIAS ? bias[gn] : 0.0f;
        float* cp = C + rowbase + gn;
        const float* rp = HAS_RES ? (resid + rowbase + gn) : nullptr;
#pragma unroll
        for (int r = 0; r < 8; ++r) {
            float val = acc[t][r] + bvs;
            if (HAS_RES) val += rp[r * N];
            if (RELU)    val = fmaxf(val, 0.0f);
            cp[r * N] = val;
        }
    }
}

// ---------------------------------------------------------------------------
// Fused flash attention, transposed-score scheme.
//   S^T = K @ Q^T (same LDS fragment reads, swapped WMMA roles) puts half of the 64
//   score values of one query row in each of lanes l, l^16; the S^T accumulator VGPRs are
//   exactly the A-fragment layout needed for P@V: no LDS P staging.
//   Q,K,V,O: [B*256, 384] f32, head h at cols [h*64, h*64+64); grid (4,H,B).
// ---------------------------------------------------------------------------
template<int CAUSAL>
__global__ __launch_bounds__(128) void attention_wmma_k(
    const float* __restrict__ Q, const float* __restrict__ Kx,
    const float* __restrict__ V, float* __restrict__ O)
{
    __shared__ _Float16 Qs[64][72];   // [m][d]
    __shared__ _Float16 Ks[64][72];   // [s][d]
    __shared__ _Float16 Vs[64][72];   // [d][s]

    const int tid  = threadIdx.x;
    const int wave = tid >> 5;
    const int lane = tid & 31;
    const int ln15 = lane & 15;
    const int hi   = (lane >> 4) & 1;

    const int b    = blockIdx.z;
    const int h    = blockIdx.y;
    const int tblk = blockIdx.x;
    const int col0 = h * 64;

    const size_t qbase  = ((size_t)b * 256 + tblk * 64) * 384 + col0;
    const size_t kvbase = (size_t)b * 256 * 384 + col0;
    const float  scale  = 0.051031036307982884f;   // 384^-0.5

    // stage Q block (pre-scaled): 8 float4 per thread
#pragma unroll
    for (int i = 0; i < 8; ++i) {
        int e = i * 128 + tid;
        int r = e >> 4, d4 = e & 15;
        v4f qv = *(const v4f*)(Q + qbase + (size_t)r * 384 + d4 * 4);
        v4h hv;
#pragma unroll
        for (int j = 0; j < 4; ++j) hv[j] = (_Float16)(qv[j] * scale);
        *(v4h*)&Qs[r][d4 * 4] = hv;
    }

    // K/V staging map: two 4x4 (s x d) patches per thread
    const int pd = tid & 15;          // d-patch
    const int ps = tid >> 4;          // s-patch (0..7); second patch ps+8
    const float* kP[2];
    const float* vP[2];
#pragma unroll
    for (int pp = 0; pp < 2; ++pp) {
        int s0 = (ps + pp * 8) * 4;
        kP[pp] = Kx + kvbase + (size_t)s0 * 384 + pd * 4;
        vP[pp] = V  + kvbase + (size_t)s0 * 384 + pd * 4;
    }

    v8f oacc[4];
#pragma unroll
    for (int t = 0; t < 4; ++t) oacc[t] = zero_v8f();
    float mrun = -3.0e38f, lrun = 0.0f;

    const int nchunk = CAUSAL ? (tblk + 1) : 4;
    for (int sc = 0; sc < nchunk; ++sc) {
        // batched K/V loads for this chunk
        v4f kq[2][4], vq[2][4];
#pragma unroll
        for (int pp = 0; pp < 2; ++pp)
#pragma unroll
            for (int i = 0; i < 4; ++i) {
                kq[pp][i] = *(const v4f*)(kP[pp] + i * 384);
                vq[pp][i] = *(const v4f*)(vP[pp] + i * 384);
            }
#pragma unroll
        for (int pp = 0; pp < 2; ++pp) {
            kP[pp] += 64 * 384;
            vP[pp] += 64 * 384;
        }
        __builtin_prefetch(kP[0], 0, 1);
        __builtin_prefetch(vP[0], 0, 1);

        __syncthreads();
#pragma unroll
        for (int pp = 0; pp < 2; ++pp) {
            int s0 = (ps + pp * 8) * 4;
            int d0 = pd * 4;
#pragma unroll
            for (int i = 0; i < 4; ++i) {
                v4h hk;
#pragma unroll
                for (int j = 0; j < 4; ++j) hk[j] = (_Float16)kq[pp][i][j];
                *(v4h*)&Ks[s0 + i][d0] = hk;
            }
#pragma unroll
            for (int j = 0; j < 4; ++j) { // 4x4 transpose for Vs[d][s]
                v4h hv;
#pragma unroll
                for (int i = 0; i < 4; ++i) hv[i] = (_Float16)vq[pp][i][j];
                *(v4h*)&Vs[d0 + j][s0] = hv;
            }
        }
        __syncthreads();

        // S^T = K @ Q^T : wave computes 64(s) x 16(m) in 4 tiles
        v8f st[4];
#pragma unroll
        for (int t = 0; t < 4; ++t) st[t] = zero_v8f();
#pragma unroll
        for (int kk = 0; kk < 64; kk += 32) {
            const int kb = kk + hi * 8;
            F16x16 bq, ak[4];
            bq.h[0] = *(const v8h*)&Qs[wave * 16 + ln15][kb];
            bq.h[1] = *(const v8h*)&Qs[wave * 16 + ln15][kb + 16];
#pragma unroll
            for (int t = 0; t < 4; ++t) {
                ak[t].h[0] = *(const v8h*)&Ks[t * 16 + ln15][kb];
                ak[t].h[1] = *(const v8h*)&Ks[t * 16 + ln15][kb + 16];
            }
#pragma unroll
            for (int t = 0; t < 4; ++t)
                st[t] = __builtin_amdgcn_wmma_f32_16x16x32_f16(
                    false, ak[t].v, false, bq.v, (short)0, st[t], false, false);
        }

        // causal mask on the diagonal chunk; lane owns row m = wave*16+ln15
        if (CAUSAL && sc == tblk) {
            const int ml = wave * 16 + ln15;
#pragma unroll
            for (int t = 0; t < 4; ++t) {
#pragma unroll
                for (int r = 0; r < 8; ++r) {
                    int sl = 16 * t + r + 8 * hi;
                    if (sl > ml) st[t][r] = -1.0e30f;
                }
            }
        }

        // online softmax: 32 of the row's 64 scores in-lane, rest in lane^16
        float nm = mrun;
#pragma unroll
        for (int t = 0; t < 4; ++t)
#pragma unroll
            for (int r = 0; r < 8; ++r) nm = fmaxf(nm, st[t][r]);
        nm = fmaxf(nm, __shfl_xor(nm, 16, 32));
        const float sc_old = __expf(mrun - nm);
        float rsum = 0.0f;
#pragma unroll
        for (int t = 0; t < 4; ++t)
#pragma unroll
            for (int r = 0; r < 8; ++r) {
                float p = __expf(st[t][r] - nm);
                st[t][r] = p;
                rsum += p;
            }
        rsum += __shfl_xor(rsum, 16, 32);
        lrun = lrun * sc_old + rsum;
        mrun = nm;

        // rescale O rows: row m = r + 8*hi stats live in lane (r + 8*hi)
#pragma unroll
        for (int r = 0; r < 8; ++r) {
            float os = __shfl(sc_old, r + 8 * hi, 32);
#pragma unroll
            for (int t = 0; t < 4; ++t) oacc[t][r] *= os;
        }

        // P@V: A-fragments come straight from st accumulators (f32->f16)
#pragma unroll
        for (int c = 0; c < 2; ++c) {
            F16x16 ap;
#pragma unroll
            for (int j = 0; j < 8; ++j) {
                ap.h[0][j] = (_Float16)st[2 * c][j];
                ap.h[1][j] = (_Float16)st[2 * c + 1][j];
            }
            const int kb = 32 * c + hi * 8;
            F16x16 bv4[4];
#pragma unroll
            for (int t = 0; t < 4; ++t) {
                bv4[t].h[0] = *(const v8h*)&Vs[t * 16 + ln15][kb];
                bv4[t].h[1] = *(const v8h*)&Vs[t * 16 + ln15][kb + 16];
            }
#pragma unroll
            for (int t = 0; t < 4; ++t)
                oacc[t] = __builtin_amdgcn_wmma_f32_16x16x32_f16(
                    false, ap.v, false, bv4[t].v, (short)0, oacc[t], false, false);
        }
    }

    // normalize + write head output (constant offsets off one base pointer)
    const float linv = 1.0f / lrun;   // for row m = ln15
    float* ob = O + qbase + (size_t)(wave * 16 + hi * 8) * 384 + ln15;
#pragma unroll
    for (int r = 0; r < 8; ++r) {
        const float li = __shfl(linv, r + 8 * hi, 32);
#pragma unroll
        for (int t = 0; t < 4; ++t)
            ob[r * 384 + t * 16] = oacc[t][r] * li;
    }
}

// ---------------------------------------------------------------------------
// Orchestration
// ---------------------------------------------------------------------------
extern "C" void kernel_launch(void* const* d_in, const int* in_sizes, int n_in,
                              void* d_out, int out_size, void* d_ws, size_t ws_size,
                              hipStream_t stream) {
    (void)in_sizes; (void)n_in; (void)out_size; (void)ws_size;

    const float* x       = (const float*)d_in[0];
    const float* enc     = (const float*)d_in[1];
    const float* wq1     = (const float*)d_in[2];
    const float* wk1     = (const float*)d_in[3];
    const float* wv1     = (const float*)d_in[4];
    const float* proj1_w = (const float*)d_in[5];
    const float* proj1_b = (const float*)d_in[6];
    const float* wq2     = (const float*)d_in[7];
    const float* wk2     = (const float*)d_in[8];
    const float* wv2     = (const float*)d_in[9];
    const float* proj2_w = (const float*)d_in[10];
    const float* proj2_b = (const float*)d_in[11];
    const float* ffn_w1  = (const float*)d_in[12];
    const float* ffn_b1  = (const float*)d_in[13];
    const float* ffn_w2  = (const float*)d_in[14];
    const float* ffn_b2  = (const float*)d_in[15];
    const float* ln1_g   = (const float*)d_in[16];
    const float* ln1_b   = (const float*)d_in[17];
    const float* ln2_g   = (const float*)d_in[18];
    const float* ln2_b   = (const float*)d_in[19];
    const float* ln3_g   = (const float*)d_in[20];
    const float* ln3_b   = (const float*)d_in[21];

    const int M = 128 * 256;             // 32768 rows
    const size_t AS = (size_t)M * 384;

    float* ws    = (float*)d_ws;
    float* lnbuf = ws;
    float* qb    = ws + 1 * AS;
    float* kb    = ws + 2 * AS;
    float* vb    = ws + 3 * AS;
    float* att   = ws + 4 * AS;
    float* x2    = ws + 5 * AS;
    float* x3    = ws + 6 * AS;
    float* ffh   = ws + 7 * AS;          // M x 1536

    dim3 blk(128);
    dim3 gLN(M);
    dim3 gG384(M / 64, 384 / 64);
    dim3 gG1536(M / 64, 1536 / 64);
    dim3 gAtt(4, 6, 128);

    // GEMM template configs:
    //   headed QKV:  N=384, K=384, BSK=64,  BSH=24576, no bias/res/relu
    //   proj+res:    N=384, K=384, BSK=384, BSH=64,    bias+res
    //   ffn1+relu:   N=1536,K=384, BSK=1536,BSH=64,    bias+relu
    //   ffn2+res:    N=384, K=1536,BSK=384, BSH=64,    bias+res

    // ---- block 1: x = x + selfattn(ln1(x)) (causal) ----
    layernorm_k<<<gLN, blk, 0, stream>>>(x, ln1_g, ln1_b, lnbuf);
    gemm_wmma_k<384, 384, 64, 24576, 0, 0, 0><<<gG384, blk, 0, stream>>>(
        lnbuf, wq1, nullptr, nullptr, qb);
    gemm_wmma_k<384, 384, 64, 24576, 0, 0, 0><<<gG384, blk, 0, stream>>>(
        lnbuf, wk1, nullptr, nullptr, kb);
    gemm_wmma_k<384, 384, 64, 24576, 0, 0, 0><<<gG384, blk, 0, stream>>>(
        lnbuf, wv1, nullptr, nullptr, vb);
    attention_wmma_k<1><<<gAtt, blk, 0, stream>>>(qb, kb, vb, att);
    gemm_wmma_k<384, 384, 384, 64, 1, 1, 0><<<gG384, blk, 0, stream>>>(
        att, proj1_w, proj1_b, x, x2);

    // ---- block 2: x = x + crossattn(ln2(x), enc) ----
    layernorm_k<<<gLN, blk, 0, stream>>>(x2, ln2_g, ln2_b, lnbuf);
    gemm_wmma_k<384, 384, 64, 24576, 0, 0, 0><<<gG384, blk, 0, stream>>>(
        lnbuf, wq2, nullptr, nullptr, qb);
    gemm_wmma_k<384, 384, 64, 24576, 0, 0, 0><<<gG384, blk, 0, stream>>>(
        enc, wk2, nullptr, nullptr, kb);
    gemm_wmma_k<384, 384, 64, 24576, 0, 0, 0><<<gG384, blk, 0, stream>>>(
        enc, wv2, nullptr, nullptr, vb);
    attention_wmma_k<0><<<gAtt, blk, 0, stream>>>(qb, kb, vb, att);
    gemm_wmma_k<384, 384, 384, 64, 1, 1, 0><<<gG384, blk, 0, stream>>>(
        att, proj2_w, proj2_b, x2, x3);

    // ---- block 3: x = x + ffn(ln3(x)) ----
    layernorm_k<<<gLN, blk, 0, stream>>>(x3, ln3_g, ln3_b, lnbuf);
    gemm_wmma_k<1536, 384, 1536, 64, 1, 0, 1><<<gG1536, blk, 0, stream>>>(
        lnbuf, ffn_w1, ffn_b1, nullptr, ffh);
    gemm_wmma_k<384, 1536, 384, 64, 1, 1, 0><<<gG384, blk, 0, stream>>>(
        ffh, ffn_w2, ffn_b2, x3, (float*)d_out);
}
